// GCN_type1_28346784153910
// MI455X (gfx1250) — matrix operand order repaired and checked by reading, not compiled, not run on hardware
//
#include <hip/hip_runtime.h>

typedef __attribute__((ext_vector_type(16))) _Float16 v16h;
typedef __attribute__((ext_vector_type(8)))  float    v8f;

#define NN 50000
#define NE 800000

// ---------------------------------------------------------------------------
// Pack f32 [M,K] row-major -> f16 A-fragments for V_WMMA_F32_16X16X32_F16.
// Fragment layout: frag[((rt*(K/32)+kt)*32 + lane)*16 + i]
//   lane 0-15  (row = rt*16+lane):   i=0..7 -> K=kt*32+{0..7},  i=8..15 -> K=kt*32+{16..23}
//   lane 16-31 (row = rt*16+lane-16):i=0..7 -> K=kt*32+{8..15}, i=8..15 -> K=kt*32+{24..31}
__global__ void pack_a(const float* __restrict__ s, _Float16* __restrict__ d,
                       int M, int K, int n) {
    int idx = blockIdx.x * blockDim.x + threadIdx.x;
    if (idx >= n) return;
    int i    = idx & 15;
    int lane = (idx >> 4) & 31;
    int t    = idx >> 9;
    int KT   = K >> 5;
    int kt   = t % KT;
    int rt   = t / KT;
    int row  = rt * 16 + (lane & 15);
    int k    = kt * 32 + (lane >> 4) * 8 + (i & 7) + (i >> 3) * 16;
    d[idx] = (_Float16)s[(size_t)row * K + k];
}

// Pack f32 weight [K,F] row-major -> f16 B-fragments (zero-pad cols F..NC).
// Fragment layout: frag[((jt*(K/32)+kt)*32 + lane)*16 + i]
//   column n = jt*16 + (lane&15); K = kt*32 + (lane>=16 ? 8 : 0) + (i&7) + (i>=8 ? 16 : 0)
__global__ void pack_b(const float* __restrict__ s, _Float16* __restrict__ d,
                       int K, int F, int NC, int n) {
    int idx = blockIdx.x * blockDim.x + threadIdx.x;
    if (idx >= n) return;
    int i    = idx & 15;
    int lane = (idx >> 4) & 31;
    int t    = idx >> 9;
    int KT   = K >> 5;
    int kt   = t % KT;
    int jt   = t / KT;
    int col  = jt * 16 + (lane & 15);
    int k    = kt * 32 + (lane >> 4) * 8 + (i & 7) + (i >> 3) * 16;
    d[idx] = (col < F) ? (_Float16)s[(size_t)k * F + col] : (_Float16)0.f;
}

// ---------------------------------------------------------------------------
// C[M,NC](f32,row-major) = A(packed frags) * B(packed frags)
// One wave per 16x16 output tile; blockIdx.x = row tile, wave = column tile.
// Inner loop per ktile: one v16h load (A) + one v16h load (B) + one v_wmma.
template <int K, int NC>
__global__ __launch_bounds__(32 * (NC / 16))
void wmma_gemm(const _Float16* __restrict__ A, const _Float16* __restrict__ B,
               float* __restrict__ C) {
    constexpr int KT = K / 32;
    const int lane = threadIdx.x & 31;
    const int wave = threadIdx.x >> 5;
    const int rt   = blockIdx.x;

    const _Float16* __restrict__ Af = A + ((size_t)rt * KT * 32 + lane) * 16;
    const _Float16* __restrict__ Bf = B + ((size_t)wave * KT * 32 + lane) * 16;

    v8f acc = {};
#pragma unroll
    for (int kt = 0; kt < KT; ++kt) {
        v16h a = *(const v16h*)(Af + (size_t)kt * 512);
        v16h b = *(const v16h*)(Bf + (size_t)kt * 512);
        acc = __builtin_amdgcn_wmma_f32_16x16x32_f16(false, a, false, b,
                                                     (short)0, acc, false, false);
    }

    // C/D layout: VGPR g -> M = g (lanes 0-15) / g+8 (lanes 16-31); N = lane&15
    const int m0   = rt * 16 + ((lane >> 4) & 1) * 8;
    const int ncol = wave * 16 + (lane & 15);
#pragma unroll
    for (int g = 0; g < 8; ++g)
        C[(size_t)(m0 + g) * NC + ncol] = acc[g];
}

// ---------------------------------------------------------------------------
// out[dst] += w * feat[src]   (one thread per edge x float4-chunk)
template <int F4>
__global__ void spmm(const int* __restrict__ src, const int* __restrict__ dst,
                     const float* __restrict__ w, const float4* __restrict__ feat,
                     float* __restrict__ out, int E) {
    int i = blockIdx.x * blockDim.x + threadIdx.x;
    if (i >= E * F4) return;
    int e = i / F4;
    int c = i - e * F4;
    float  ww = w[e];
    float4 v  = feat[(size_t)src[e] * F4 + c];
    float* o  = out + (size_t)dst[e] * (F4 * 4) + c * 4;
    atomicAdd(o + 0, ww * v.x);
    atomicAdd(o + 1, ww * v.y);
    atomicAdd(o + 2, ww * v.z);
    atomicAdd(o + 3, ww * v.w);
}

// ---------------------------------------------------------------------------
// per-column sum / sum-of-squares; blockDim.x = ld (>= F); 64 rows per block
__global__ void colstats(const float* __restrict__ h, int n, int F, int ld,
                         float* __restrict__ sums, float* __restrict__ sumsq) {
    int c = threadIdx.x;
    if (c >= F) return;
    int r0   = blockIdx.x * 64;
    int rend = r0 + 64;
    if (rend > n) rend = n;
    float s = 0.f, q = 0.f;
    for (int r = r0; r < rend; ++r) {
        __builtin_prefetch(&h[(size_t)(r + 16) * ld + c], 0, 3);
        float v = h[(size_t)r * ld + c];
        s += v;
        q += v * v;
    }
    atomicAdd(&sums[c], s);
    atomicAdd(&sumsq[c], q);
}

// BN (batch stats, biased var) + leaky_relu(0.01); writes f16 directly in
// packed-A fragment layout so the next WMMA GEMM loads it as v16h fragments.
template <int F>
__global__ void bn_leaky_pack(const float* __restrict__ h, const float* __restrict__ sums,
                              const float* __restrict__ sumsq, _Float16* __restrict__ out,
                              int n) {
    int i = blockIdx.x * blockDim.x + threadIdx.x;
    if (i >= n) return;
    int r = i / F;
    int c = i - r * F;
    float mean = sums[c] * (1.f / NN);
    float var  = sumsq[c] * (1.f / NN) - mean * mean;
    float sc   = rsqrtf(var + 1e-5f);
    float y    = (h[i] - mean) * sc;
    y = (y > 0.f) ? y : 0.01f * y;

    // (r,c) -> packed A-fragment index
    int rt = r >> 4, rl = r & 15;
    int kt = c >> 5, rem = c & 31;
    int hs  = (rem >> 3) & 1;
    int i16 = (rem & 7) + ((rem >> 4) & 1) * 8;
    int lane = hs * 16 + rl;
    size_t oidx = ((size_t)(rt * (F / 32) + kt) * 32 + lane) * 16 + i16;
    out[oidx] = (_Float16)y;
}

// final BN, padded (ld=32) f32 input, 20 valid cols, f32 output, no activation
__global__ void bn_final(const float* __restrict__ h, const float* __restrict__ sums,
                         const float* __restrict__ sumsq, float* __restrict__ out, int n) {
    int i = blockIdx.x * blockDim.x + threadIdx.x;
    if (i >= n) return;
    int r = i / 20;
    int c = i - r * 20;
    float mean = sums[c] * (1.f / NN);
    float var  = sumsq[c] * (1.f / NN) - mean * mean;
    float sc   = rsqrtf(var + 1e-5f);
    out[i] = (h[(size_t)r * 32 + c] - mean) * sc;
}

// ---------------------------------------------------------------------------
extern "C" void kernel_launch(void* const* d_in, const int* in_sizes, int n_in,
                              void* d_out, int out_size, void* d_ws, size_t ws_size,
                              hipStream_t stream) {
    const float* x      = (const float*)d_in[0];
    const float* edge_w = (const float*)d_in[1];
    const float* W1     = (const float*)d_in[2];
    // d_in[3] = b1 : constant column shift before BN -> cancels, unused
    const float* W2     = (const float*)d_in[4];
    // d_in[5] = b2 : cancels, unused
    const float* Wl     = (const float*)d_in[6];
    // d_in[7] = bl : cancels, unused
    const int* esrc = (const int*)d_in[8];
    const int* edst = (const int*)d_in[9];
    float* out = (float*)d_out;

    // workspace layout (bytes); A/B/C regions reused across stages
    char* ws = (char*)d_ws;
    const size_t OFF_A = 0;                  // 25.6 MB: xa -> h1n -> h2n (packed f16)
    const size_t OFF_B = 25600000;           // 25.6 MB: h1a -> h2a -> h3(padded)
    const size_t OFF_C = 51200000;           // 25.6 MB: h1 -> h2
    const size_t OFF_W = 76800000;           // packed weights (f16) + stats

    _Float16* xa  = (_Float16*)(ws + OFF_A);
    float*    h1a = (float*)(ws + OFF_B);
    float*    h1  = (float*)(ws + OFF_C);
    _Float16* h1n = (_Float16*)(ws + OFF_A);
    float*    h2a = (float*)(ws + OFF_B);
    float*    h2  = (float*)(ws + OFF_C);
    _Float16* h2n = (_Float16*)(ws + OFF_A);
    float*    h3  = (float*)(ws + OFF_B);

    _Float16* w1p = (_Float16*)(ws + OFF_W);            // 256*128*2 = 64 KB
    _Float16* w2p = (_Float16*)(ws + OFF_W + 65536);    // 128*64*2  = 16 KB
    _Float16* wlp = (_Float16*)(ws + OFF_W + 81920);    // 64*32*2   = 4 KB (padded)
    float*    sums  = (float*)(ws + OFF_W + 86016);     // 128 f32
    float*    sumsq = sums + 128;                       // 128 f32

    // ---- stage 0: pack operands into WMMA fragment layout ------------------
    pack_a<<<(NN * 256 + 255) / 256, 256, 0, stream>>>(x, xa, NN, 256, NN * 256);
    pack_b<<<(256 * 128 + 255) / 256, 256, 0, stream>>>(W1, w1p, 256, 128, 128, 256 * 128);
    pack_b<<<(128 * 64 + 255) / 256, 256, 0, stream>>>(W2, w2p, 128, 64, 64, 128 * 64);
    pack_b<<<(64 * 32 + 255) / 256, 256, 0, stream>>>(Wl, wlp, 64, 20, 32, 64 * 32);

    // ---- layer 1: x @ W1 -> spmm -> BN+leaky -------------------------------
    wmma_gemm<256, 128><<<NN / 16, 256, 0, stream>>>(xa, w1p, h1a);
    hipMemsetAsync(h1, 0, (size_t)NN * 128 * 4, stream);
    spmm<32><<<(NE * 32 + 255) / 256, 256, 0, stream>>>(esrc, edst, edge_w,
                                                        (const float4*)h1a, h1, NE);
    hipMemsetAsync(sums, 0, 1024, stream);
    colstats<<<(NN + 63) / 64, 128, 0, stream>>>(h1, NN, 128, 128, sums, sumsq);
    bn_leaky_pack<128><<<(NN * 128 + 255) / 256, 256, 0, stream>>>(h1, sums, sumsq,
                                                                   h1n, NN * 128);

    // ---- layer 2: h @ W2 -> spmm -> BN+leaky -------------------------------
    wmma_gemm<128, 64><<<NN / 16, 128, 0, stream>>>(h1n, w2p, h2a);
    hipMemsetAsync(h2, 0, (size_t)NN * 64 * 4, stream);
    spmm<16><<<(NE * 16 + 255) / 256, 256, 0, stream>>>(esrc, edst, edge_w,
                                                        (const float4*)h2a, h2, NE);
    hipMemsetAsync(sums, 0, 1024, stream);
    colstats<<<(NN + 63) / 64, 64, 0, stream>>>(h2, NN, 64, 64, sums, sumsq);
    bn_leaky_pack<64><<<(NN * 64 + 255) / 256, 256, 0, stream>>>(h2, sums, sumsq,
                                                                 h2n, NN * 64);

    // ---- head: h @ Wl (padded to 32 cols) -> BN ----------------------------
    wmma_gemm<64, 32><<<NN / 16, 64, 0, stream>>>(h2n, wlp, h3);
    hipMemsetAsync(sums, 0, 1024, stream);
    colstats<<<(NN + 63) / 64, 32, 0, stream>>>(h3, NN, 20, 32, sums, sumsq);
    bn_final<<<(NN * 20 + 255) / 256, 256, 0, stream>>>(h3, sums, sumsq, out, NN * 20);
}
